// GRUD_cells_20529943674896
// MI455X (gfx1250) — matrix operand order repaired and checked by reading, not compiled, not run on hardware
//
#include <hip/hip_runtime.h>
#include <hip/hip_bf16.h>
#include <math.h>

// Problem dims (fixed by the reference)
constexpr int NB = 256;   // batch
constexpr int NT = 512;   // time steps
constexpr int ND = 64;    // input dim
constexpr int NH = 128;   // hidden dim

typedef __attribute__((ext_vector_type(16))) __bf16 v16bf;
typedef __attribute__((ext_vector_type(8)))  __bf16 v8bf;
typedef __attribute__((ext_vector_type(8)))  float  v8f;

// Build a 16-wide bf16 WMMA operand from two 8-element (16-byte) LDS chunks.
__device__ __forceinline__ v16bf ld16(const __bf16* p0, const __bf16* p1) {
  v8bf lo = *(const v8bf*)p0;
  v8bf hi = *(const v8bf*)p1;
  return __builtin_shufflevector(lo, hi, 0, 1, 2, 3, 4, 5, 6, 7,
                                         8, 9, 10, 11, 12, 13, 14, 15);
}

__device__ __forceinline__ v8f wmma_bf16(v16bf a, v16bf b, v8f c) {
  // (neg_a, A, neg_b, B, c_mod, C, reuse_a, reuse_b)
  return __builtin_amdgcn_wmma_f32_16x16x32_bf16(false, a, false, b,
                                                 (short)0, c, false, false);
}

__device__ __forceinline__ v8f splat8(float x) {
  v8f v = {x, x, x, x, x, x, x, x};
  return v;
}

__device__ __forceinline__ float sigmoid_fast(float x) {
  return 1.0f / (1.0f + __expf(-x));
}

// ---------------------------------------------------------------------------
// Phase 1: h-independent work. Computes, for every (b,t):
//   pre_g = dlt @ w_gamma_h + b_gamma_h
//   pre_z = x @ w_zx + m @ w_zm + b_z        (x = decay-imputed input)
//   pre_r = x @ w_rx + m @ w_rm + b_r
//   pre_h = x @ w_hx + m @ w_hm + b_h
// stored as [T][B][H] fp32 so the scan kernel reads contiguous chunks.
// ---------------------------------------------------------------------------
struct P1Args {
  const float *values, *masks, *deltas, *emp_mean, *xlocf;
  const float *w_gamma_x, *b_gamma_x;
  const float *w_gamma_h, *w_zx, *w_zm, *w_rx, *w_rm, *w_hx, *w_hm;
  const float *b_gamma_h, *b_z, *b_r, *b_h;
  float *pre_g, *pre_z, *pre_r, *pre_h;
};

__global__ __launch_bounds__(256) void grud_precompute(P1Args A) {
  extern __shared__ __bf16 smem[];
  // 7 transposed weights Wt[n][k] (bf16), then x/m/dlt row tiles [16][ND]
  __bf16* Wg  = smem;
  __bf16* Wzx = Wg  + NH * ND;
  __bf16* Wzm = Wzx + NH * ND;
  __bf16* Wrx = Wzm + NH * ND;
  __bf16* Wrm = Wrx + NH * ND;
  __bf16* Whx = Wrm + NH * ND;
  __bf16* Whm = Whx + NH * ND;
  __bf16* xs  = Whm + NH * ND;
  __bf16* ms  = xs + 16 * ND;
  __bf16* dls = ms + 16 * ND;

  const int tid = threadIdx.x;

  // Load + transpose all input-side weights into LDS (bf16).
  for (int i = tid; i < NH * ND; i += 256) {
    const int n = i / ND, k = i % ND;
    const int src = k * NH + n;
    Wg[i]  = (__bf16)A.w_gamma_h[src];
    Wzx[i] = (__bf16)A.w_zx[src];
    Wzm[i] = (__bf16)A.w_zm[src];
    Wrx[i] = (__bf16)A.w_rx[src];
    Wrm[i] = (__bf16)A.w_rm[src];
    Whx[i] = (__bf16)A.w_hx[src];
    Whm[i] = (__bf16)A.w_hm[src];
  }
  __syncthreads();

  const int lane  = tid & 31;
  const int wv    = tid >> 5;      // wave id == N-tile (H/16 == 8 tiles)
  const int ln    = lane & 15;
  const int khalf = lane >> 4;
  const int ng    = wv * 16 + ln;  // global output column

  const float bg = A.b_gamma_h[ng];
  const float bz = A.b_z[ng];
  const float br = A.b_r[ng];
  const float bh = A.b_h[ng];

  const int NTILES = NB * NT / 16;  // 8192 row tiles of 16 (b,t) rows
  for (int rt = blockIdx.x; rt < NTILES; rt += gridDim.x) {
    const int r0 = rt * 16;  // flat row = b*NT + t ; T%16==0 so b is constant

    // Elementwise: build x (imputed), m, dlt tiles in LDS as bf16.
    for (int i = tid; i < 16 * ND; i += 256) {
      const int mrow = i >> 6;     // /ND
      const int d    = i & (ND - 1);
      const long idx = (long)(r0 + mrow) * ND + d;
      const float val = A.values[idx];
      const float msk = A.masks[idx];
      const float dl  = A.deltas[idx];
      const float xl  = A.xlocf[idx];
      const float gx  = __expf(-fmaxf(A.w_gamma_x[d] * dl + A.b_gamma_x[d], 0.0f));
      const float x   = msk * val +
                        (1.0f - msk) * (gx * xl + (1.0f - gx) * A.emp_mean[d]);
      xs[i]  = (__bf16)x;
      ms[i]  = (__bf16)msk;
      dls[i] = (__bf16)dl;
    }
    __syncthreads();

    v8f ag = splat8(bg), az = splat8(bz), ar = splat8(br), ah = splat8(bh);

    const __bf16* arow_x = xs  + ln * ND;
    const __bf16* arow_m = ms  + ln * ND;
    const __bf16* arow_d = dls + ln * ND;

#pragma unroll
    for (int kk = 0; kk < 2; ++kk) {            // K = 64 -> 2 chunks of 32
      const int k0 = kk * 32;
      const int ao = k0 + khalf * 16;
      const v16bf ax = ld16(arow_x + ao, arow_x + ao + 8);
      const v16bf am = ld16(arow_m + ao, arow_m + ao + 8);
      const v16bf ad = ld16(arow_d + ao, arow_d + ao + 8);
      const int bo = ng * ND + k0 + khalf * 16;
      ag = wmma_bf16(ad, ld16(Wg  + bo, Wg  + bo + 8), ag);
      az = wmma_bf16(ax, ld16(Wzx + bo, Wzx + bo + 8), az);
      az = wmma_bf16(am, ld16(Wzm + bo, Wzm + bo + 8), az);
      ar = wmma_bf16(ax, ld16(Wrx + bo, Wrx + bo + 8), ar);
      ar = wmma_bf16(am, ld16(Wrm + bo, Wrm + bo + 8), ar);
      ah = wmma_bf16(ax, ld16(Whx + bo, Whx + bo + 8), ah);
      ah = wmma_bf16(am, ld16(Whm + bo, Whm + bo + 8), ah);
    }

    // Store accumulators to [T][B][H] workspace (C layout: m = v + khalf*8).
    const int bglob = r0 >> 9;       // / NT
    const int t0    = r0 & (NT - 1);
#pragma unroll
    for (int v = 0; v < 8; ++v) {
      const int m = v + khalf * 8;
      const long o = ((long)(t0 + m) * NB + bglob) * NH + ng;
      A.pre_g[o] = ag[v];
      A.pre_z[o] = az[v];
      A.pre_r[o] = ar[v];
      A.pre_h[o] = ah[v];
    }
    __syncthreads();  // protect LDS tiles before next row tile
  }
}

// ---------------------------------------------------------------------------
// Phase 2: sequential scan. Batch rows are independent -> 16 blocks each own
// 16 rows of h with zero inter-block sync. Recurrent-weight B-fragments are
// loop-invariant per wave and live in VGPRs (96 regs); h lives in f32
// registers (C-layout ownership) + a tiny bf16 LDS mirror for the WMMA
// A-matrix. Next step's pre-activations are prefetched to hide HBM latency.
// ---------------------------------------------------------------------------
struct P2Args {
  const float *w_zh, *w_rh, *w_hh;
  const float *pre_g, *pre_z, *pre_r, *pre_h;
  float *out_hidden;  // [B][T][H]
  float *out_hlast;   // [B][H]
};

__global__ __launch_bounds__(256) void grud_scan(P2Args A) {
  __shared__ __bf16 hlds[16 * NH];    // bf16 mirror of decayed h
  __shared__ __bf16 rhlds[16 * NH];   // bf16 r*h

  const int tid    = threadIdx.x;
  const int lane   = tid & 31;
  const int wv     = tid >> 5;
  const int ln     = lane & 15;
  const int khalf  = lane >> 4;
  const int ng     = wv * 16 + ln;
  const int bchunk = blockIdx.x;      // 0..15

  // Preload this wave's loop-invariant B-fragments straight from global,
  // applying the K-major layout: b[i] = W[(k0 + khalf*8 + (i<8 ? i : 8+i))*NH + ng].
  v16bf Bz[4], Br[4], Bh[4];
#pragma unroll
  for (int kk = 0; kk < 4; ++kk) {
#pragma unroll
    for (int i = 0; i < 16; ++i) {
      const int k = kk * 32 + khalf * 8 + (i < 8 ? i : 8 + i);
      Bz[kk][i] = (__bf16)A.w_zh[k * NH + ng];
      Br[kk][i] = (__bf16)A.w_rh[k * NH + ng];
      Bh[kk][i] = (__bf16)A.w_hh[k * NH + ng];
    }
  }

  float hreg[8];
#pragma unroll
  for (int v = 0; v < 8; ++v) hreg[v] = 0.0f;

  // Prime the software pipeline with step 0's pre-activations.
  float g[8], pz[8], pr[8], ph[8];
  {
    const long base0 = ((long)0 * NB + bchunk * 16) * NH;
#pragma unroll
    for (int v = 0; v < 8; ++v) {
      const int m = v + khalf * 8;
      g[v]  = A.pre_g[base0 + m * NH + ng];
      pz[v] = A.pre_z[base0 + m * NH + ng];
      pr[v] = A.pre_r[base0 + m * NH + ng];
      ph[v] = A.pre_h[base0 + m * NH + ng];
    }
  }

  for (int t = 0; t < NT; ++t) {
    // (1) gamma decay of h; publish bf16 h to LDS as the WMMA A-matrix.
#pragma unroll
    for (int v = 0; v < 8; ++v) {
      const int m = v + khalf * 8;
      hreg[v] *= __expf(-fmaxf(g[v], 0.0f));
      hlds[m * NH + ng] = (__bf16)hreg[v];
    }
    __syncthreads();  // barrier A

    // Prefetch step t+1 (uniform index: EXEC stays all-ones for WMMA).
    const int tp = (t + 1 < NT) ? (t + 1) : t;
    const long basep = ((long)tp * NB + bchunk * 16) * NH;
    float g2[8], pz2[8], pr2[8], ph2[8];
#pragma unroll
    for (int v = 0; v < 8; ++v) {
      const int m = v + khalf * 8;
      g2[v]  = A.pre_g[basep + m * NH + ng];
      pz2[v] = A.pre_z[basep + m * NH + ng];
      pr2[v] = A.pre_r[basep + m * NH + ng];
      ph2[v] = A.pre_h[basep + m * NH + ng];
    }

    // (2) z and r gates: pre + h @ W  (K = 128 -> 4 chunks of 32).
    v8f az, ar;
#pragma unroll
    for (int v = 0; v < 8; ++v) { az[v] = pz[v]; ar[v] = pr[v]; }
    const __bf16* arow_h = hlds + ln * NH;
#pragma unroll
    for (int kk = 0; kk < 4; ++kk) {
      const int ao = kk * 32 + khalf * 8;
      const v16bf a = ld16(arow_h + ao, arow_h + ao + 16);
      az = wmma_bf16(a, Bz[kk], az);
      ar = wmma_bf16(a, Br[kk], ar);
    }

    float zv[8];
#pragma unroll
    for (int v = 0; v < 8; ++v) {
      const int m = v + khalf * 8;
      const float z = sigmoid_fast(az[v]);
      const float r = sigmoid_fast(ar[v]);
      zv[v] = z;
      rhlds[m * NH + ng] = (__bf16)(r * hreg[v]);
    }
    __syncthreads();  // barrier B

    // (3) candidate: pre_h + (r*h) @ w_hh, then blend and stream out.
    v8f ah;
#pragma unroll
    for (int v = 0; v < 8; ++v) ah[v] = ph[v];
    const __bf16* arow_rh = rhlds + ln * NH;
#pragma unroll
    for (int kk = 0; kk < 4; ++kk) {
      const int ao = kk * 32 + khalf * 8;
      const v16bf a = ld16(arow_rh + ao, arow_rh + ao + 16);
      ah = wmma_bf16(a, Bh[kk], ah);
    }

#pragma unroll
    for (int v = 0; v < 8; ++v) {
      const int m = v + khalf * 8;
      const float z  = zv[v];
      const float hn = (1.0f - z) * hreg[v] + z * tanhf(ah[v]);
      hreg[v] = hn;
      const long bg = bchunk * 16 + m;
      A.out_hidden[(bg * (long)NT + t) * NH + ng] = hn;
    }

    // Rotate the prefetch buffers into place for the next step.
#pragma unroll
    for (int v = 0; v < 8; ++v) {
      g[v] = g2[v]; pz[v] = pz2[v]; pr[v] = pr2[v]; ph[v] = ph2[v];
    }
    // No extra barrier needed: next step's hlds writes are own-lane only and
    // all cross-wave hlds/rhlds reads are fenced by barriers A/B.
  }

#pragma unroll
  for (int v = 0; v < 8; ++v) {
    const int m = v + khalf * 8;
    A.out_hlast[(long)(bchunk * 16 + m) * NH + ng] = hreg[v];
  }
}

// ---------------------------------------------------------------------------
extern "C" void kernel_launch(void* const* d_in, const int* in_sizes, int n_in,
                              void* d_out, int out_size, void* d_ws, size_t ws_size,
                              hipStream_t stream) {
  const float* values    = (const float*)d_in[0];
  const float* masks     = (const float*)d_in[1];
  const float* deltas    = (const float*)d_in[2];
  const float* emp_mean  = (const float*)d_in[3];
  const float* xlocf     = (const float*)d_in[4];
  const float* w_gamma_x = (const float*)d_in[5];
  const float* w_gamma_h = (const float*)d_in[6];
  const float* w_rx      = (const float*)d_in[7];
  const float* w_rh      = (const float*)d_in[8];
  const float* w_rm      = (const float*)d_in[9];
  const float* w_zx      = (const float*)d_in[10];
  const float* w_zh      = (const float*)d_in[11];
  const float* w_zm      = (const float*)d_in[12];
  const float* w_hx      = (const float*)d_in[13];
  const float* w_hh      = (const float*)d_in[14];
  const float* w_hm      = (const float*)d_in[15];
  const float* b_gamma_x = (const float*)d_in[16];
  const float* b_gamma_h = (const float*)d_in[17];
  const float* b_r       = (const float*)d_in[18];
  const float* b_z       = (const float*)d_in[19];
  const float* b_h       = (const float*)d_in[20];

  // Workspace: 4 pre-activation arrays [T][B][H] fp32 (64 MB each).
  const size_t per = (size_t)NT * NB * NH;
  float* pre_g = (float*)d_ws;
  float* pre_z = pre_g + per;
  float* pre_r = pre_z + per;
  float* pre_h = pre_r + per;

  float* out_hidden = (float*)d_out;                       // [B][T][H]
  float* out_hlast  = out_hidden + (size_t)NB * NT * NH;   // [B][H]

  const size_t smem1 = (size_t)(7 * NH * ND + 3 * 16 * ND) * sizeof(__bf16);

  // Allow >64KB dynamic LDS for the precompute kernel (WGP: up to 320KB).
  (void)hipFuncSetAttribute(reinterpret_cast<const void*>(grud_precompute),
                            hipFuncAttributeMaxDynamicSharedMemorySize, (int)smem1);

  P1Args a1;
  a1.values = values; a1.masks = masks; a1.deltas = deltas;
  a1.emp_mean = emp_mean; a1.xlocf = xlocf;
  a1.w_gamma_x = w_gamma_x; a1.b_gamma_x = b_gamma_x;
  a1.w_gamma_h = w_gamma_h;
  a1.w_zx = w_zx; a1.w_zm = w_zm;
  a1.w_rx = w_rx; a1.w_rm = w_rm;
  a1.w_hx = w_hx; a1.w_hm = w_hm;
  a1.b_gamma_h = b_gamma_h; a1.b_z = b_z; a1.b_r = b_r; a1.b_h = b_h;
  a1.pre_g = pre_g; a1.pre_z = pre_z; a1.pre_r = pre_r; a1.pre_h = pre_h;
  grud_precompute<<<256, 256, smem1, stream>>>(a1);

  P2Args a2;
  a2.w_zh = w_zh; a2.w_rh = w_rh; a2.w_hh = w_hh;
  a2.pre_g = pre_g; a2.pre_z = pre_z; a2.pre_r = pre_r; a2.pre_h = pre_h;
  a2.out_hidden = out_hidden; a2.out_hlast = out_hlast;
  grud_scan<<<16, 256, 0, stream>>>(a2);
}
